// xattn_block_8108898254867
// MI455X (gfx1250) — matrix operand order repaired and checked by reading, not compiled, not run on hardware
//
#include <hip/hip_runtime.h>
#include <hip/hip_bf16.h>
#include <stdint.h>

typedef __bf16 bf16;
typedef bf16  v16bf __attribute__((ext_vector_type(16)));
typedef float v8f   __attribute__((ext_vector_type(8)));

#define WMMA_BF16(a,b,c) \
  __builtin_amdgcn_wmma_f32_16x16x32_bf16(false,(a),false,(b),(short)0,(c),false,false)

static __device__ __forceinline__ bf16 to_bf16(float f){ return (bf16)f; }

// ---- WMMA fragment loaders (CDNA5 16x16x32 bf16 layouts, ISA 7.12.2) ----
// A: 16x32 (MxK). lane l: m=l&15; VGPR j<4: K=(l>>4)*8+2j{,+1}; j>=4: K=16+(l>>4)*8+2(j-4){,+1}
static __device__ __forceinline__ v16bf load_frag_a(const bf16* base, int rs, int ks){
  int l = threadIdx.x & 31;
  int r = l & 15;
  int kh = (l >> 4) << 3;
  v16bf f;
#pragma unroll
  for (int j = 0; j < 4; ++j){
    int k  = kh + 2*j;
    int k2 = 16 + kh + 2*j;
    f[2*j]     = base[r*rs + k*ks];
    f[2*j+1]   = base[r*rs + (k+1)*ks];
    f[8+2*j]   = base[r*rs + k2*ks];
    f[8+2*j+1] = base[r*rs + (k2+1)*ks];
  }
  return f;
}
// B: 32x16 (KxN) given as B^T rows (N x K). lane l: n=l&15; VGPR j<4: K=(l>>4)*8+2j{,+1}; j>=4: K=16+(l>>4)*8+2(j-4){,+1}
static __device__ __forceinline__ v16bf load_frag_b(const bf16* base, int rs, int ks){
  int l = threadIdx.x & 31;
  int r = l & 15;
  int kh = (l >> 4) << 3;
  v16bf f;
#pragma unroll
  for (int j = 0; j < 8; ++j){
    int k = kh + 2*j + (j >= 4 ? 8 : 0);
    f[2*j]   = base[r*rs + k*ks];
    f[2*j+1] = base[r*rs + (k+1)*ks];
  }
  return f;
}

// ---- elementwise f32 -> bf16 ----
__global__ __launch_bounds__(256) void cvt_f32_bf16(const float* __restrict__ S,
                                                    bf16* __restrict__ D, int n){
  int i = blockIdx.x*256 + threadIdx.x;
  if (i < n) D[i] = to_bf16(S[i]);
}

// WkT[h][c][d] = Wkv[h*128+d][c]  (transposed K-head weights, bf16), 8*256*64 elems
__global__ __launch_bounds__(256) void build_wkT(const float* __restrict__ Wkv,
                                                 bf16* __restrict__ WkT){
  int i = blockIdx.x*256 + threadIdx.x;
  int h = i >> 14, rem = i & 16383, c = rem >> 6, d = rem & 63;
  WkT[i] = to_bf16(Wkv[((size_t)(h*128 + d))*256 + c]);
}

// ---- per-row layernorm: X (R x C) f32 -> Y bf16 ----
__global__ __launch_bounds__(256) void ln_rows(const float* __restrict__ X,
                                               bf16* __restrict__ Y, int C){
  __shared__ float rs[256], rq[256];
  int row = blockIdx.x;
  const float* x = X + (size_t)row*C;
  float s = 0.f, q = 0.f;
  for (int c = threadIdx.x; c < C; c += 256){ float v = x[c]; s += v; q += v*v; }
  rs[threadIdx.x] = s; rq[threadIdx.x] = q;
  __syncthreads();
  for (int off = 128; off; off >>= 1){
    if (threadIdx.x < off){ rs[threadIdx.x] += rs[threadIdx.x+off]; rq[threadIdx.x] += rq[threadIdx.x+off]; }
    __syncthreads();
  }
  float mu  = rs[0] / (float)C;
  float var = rq[0] / (float)C - mu*mu;
  float rsig = rsqrtf(var + 1e-5f);
  for (int c = threadIdx.x; c < C; c += 256)
    Y[(size_t)row*C + c] = to_bf16((x[c] - mu) * rsig);
}

// ---- generic bf16 WMMA GEMM: C[M,N] = A[M,K] * B^T[N,K], BM=BN=64, BK=32 ----
__global__ __launch_bounds__(256) void gemm_bf16nt(const bf16* __restrict__ A, int lda,
                                                   const bf16* __restrict__ B, int ldb,
                                                   void* __restrict__ C, int ldc, int K,
                                                   int out_bf16, const float* __restrict__ bias){
  __shared__ __align__(16) bf16 As[64*32];
  __shared__ __align__(16) bf16 Bs[64*32];
  int bm = blockIdx.y, bn = blockIdx.x;
  int wid = threadIdx.x >> 5;
  int wm = wid & 3, wn = wid >> 2;         // 4x2 wave grid: 16-row x 32-col per wave
  v8f acc0 = {}; v8f acc1 = {};
  int r  = threadIdx.x >> 2;               // 0..63
  int c8 = (threadIdx.x & 3) << 3;         // 0,8,16,24
  const bf16* Ab = A + (size_t)(bm*64 + r)*lda + c8;
  const bf16* Bb = B + (size_t)(bn*64 + r)*ldb + c8;
  for (int k0 = 0; k0 < K; k0 += 32){
    *(uint4*)&As[r*32 + c8] = *(const uint4*)(Ab + k0);
    *(uint4*)&Bs[r*32 + c8] = *(const uint4*)(Bb + k0);
    if (k0 + 32 < K){
      __builtin_prefetch(Ab + k0 + 32, 0, 0);   // global_prefetch_b8
      __builtin_prefetch(Bb + k0 + 32, 0, 0);
    }
    __syncthreads();
    v16bf a  = load_frag_a(&As[wm*16*32], 32, 1);
    v16bf b0 = load_frag_b(&Bs[(wn*32)*32], 32, 1);
    v16bf b1 = load_frag_b(&Bs[(wn*32+16)*32], 32, 1);
    acc0 = WMMA_BF16(a, b0, acc0);
    acc1 = WMMA_BF16(a, b1, acc1);
    __syncthreads();
  }
  int l  = threadIdx.x & 31;
  int nn = l & 15;
  int mb = (l >> 4) << 3;
  int grow0 = bm*64 + wm*16;
  int gcol0 = bn*64 + wn*32;
#pragma unroll
  for (int j = 0; j < 8; ++j){
    int m = grow0 + mb + j;
    if (out_bf16){
      ((bf16*)C)[(size_t)m*ldc + gcol0 + nn]      = to_bf16(acc0[j]);
      ((bf16*)C)[(size_t)m*ldc + gcol0 + 16 + nn] = to_bf16(acc1[j]);
    } else {
      float b0v = bias ? bias[gcol0 + nn]      : 0.f;
      float b1v = bias ? bias[gcol0 + 16 + nn] : 0.f;
      ((float*)C)[(size_t)m*ldc + gcol0 + nn]      = acc0[j] + b0v;
      ((float*)C)[(size_t)m*ldc + gcol0 + 16 + nn] = acc1[j] + b1v;
    }
  }
}

// ---- fused attention core: one workgroup per query row n ----
// reads subseq (64x256 f32) once, LN -> bf16 LDS; g (8x256 bf16) async to LDS;
// scores = scale * g @ LNs^T  -> silu/w -> p ; t = p @ LNs  -> global bf16
__global__ __launch_bounds__(256) void attn_core(const float* __restrict__ sub,
                                                 const bf16* __restrict__ G,
                                                 bf16* __restrict__ T){
  __shared__ __align__(16) bf16 sB[64*256]; // 32KB: LN(sub) bf16, row=w, col=c
  __shared__ __align__(16) bf16 gT[16*256]; // 8KB : g rows 0..7, zero rows 8..15
  __shared__ __align__(16) bf16 pB[16*64];  // 2KB : silu(scores)/w
  int n   = blockIdx.x;
  int tid = threadIdx.x;
  int wv  = tid >> 5, l = tid & 31;

  // async-load g[n] (8x256 bf16 = 4096B) into LDS while we do LN math
  {
    const bf16* gsrc = G + (size_t)n*2048;
    unsigned lds = (unsigned)(uintptr_t)(&gT[0]) + (unsigned)(tid*16);
    unsigned long long ga = (unsigned long long)(uintptr_t)gsrc + (unsigned long long)(tid*16);
    asm volatile("global_load_async_to_lds_b128 %0, %1, off" :: "v"(lds), "v"(ga) : "memory");
  }
  // zero pad rows 8..15 of gT
  *(float4*)&gT[2048 + tid*8] = make_float4(0.f, 0.f, 0.f, 0.f);

  // per-row LN of subseq: wave wv handles rows wv*8 .. wv*8+7 (8 floats/lane)
  {
    const float* srcn = sub + (size_t)n*16384;
#pragma unroll
    for (int i = 0; i < 8; ++i){
      int r = wv*8 + i;
      const float* rp = srcn + r*256 + l*8;
      float4 x0 = *(const float4*)rp;
      float4 x1 = *(const float4*)(rp + 4);
      float vals[8] = {x0.x, x0.y, x0.z, x0.w, x1.x, x1.y, x1.z, x1.w};
      float s = 0.f, q = 0.f;
#pragma unroll
      for (int j = 0; j < 8; ++j){ s += vals[j]; q += vals[j]*vals[j]; }
#pragma unroll
      for (int off = 16; off; off >>= 1){ s += __shfl_xor(s, off); q += __shfl_xor(q, off); }
      float mu   = s * (1.f/256.f);
      float var  = q * (1.f/256.f) - mu*mu;
      float rsig = rsqrtf(var + 1e-5f);
      bf16* dp = &sB[r*256 + l*8];
#pragma unroll
      for (int j = 0; j < 8; ++j) dp[j] = to_bf16((vals[j] - mu) * rsig);
    }
  }
  asm volatile("s_wait_asynccnt 0x0" ::: "memory");
  __syncthreads();

  // scores: (16x64) = gT(16x256) @ sB^T ; waves 0..3 take one 16-col tile of w
  if (wv < 4){
    v8f acc = {};
#pragma unroll
    for (int kk = 0; kk < 8; ++kk){
      v16bf a = load_frag_a(&gT[kk*32], 256, 1);
      v16bf b = load_frag_b(&sB[(wv*16)*256 + kk*32], 256, 1);
      acc = WMMA_BF16(a, b, acc);
    }
    int nn = l & 15, mb = (l >> 4) << 3;
#pragma unroll
    for (int j = 0; j < 8; ++j){
      float sc = acc[j] * 0.125f;                         // D^-0.5, D=64
      float p  = sc * (1.f/(1.f + __expf(-sc))) * (1.f/64.f); // silu/w
      pB[(mb + j)*64 + wv*16 + nn] = to_bf16(p);
    }
  }
  __syncthreads();

  // t: (16x256) = pB(16x64) @ sB(64x256); each wave does col-tiles wv and wv+8
#pragma unroll
  for (int ti = 0; ti < 2; ++ti){
    int nt = wv + ti*8;
    v8f acc = {};
#pragma unroll
    for (int kk = 0; kk < 2; ++kk){
      v16bf a = load_frag_a(&pB[kk*32], 64, 1);
      v16bf b = load_frag_b(&sB[kk*32*256 + nt*16], 1, 256); // B^T[c,w] = sB[w][c]
      acc = WMMA_BF16(a, b, acc);
    }
    int nn = l & 15, mb = (l >> 4) << 3;
#pragma unroll
    for (int j = 0; j < 8; ++j){
      int m = mb + j;
      if (m < 8) T[(size_t)n*2048 + m*256 + nt*16 + nn] = to_bf16(acc[j]);
    }
  }
}

// ---- z[n,512] = LN(attnout[n]) * u[n]  (u = qu[:, h*128+64+d]) ----
__global__ __launch_bounds__(256) void make_z(const float* __restrict__ attn,
                                              const bf16* __restrict__ qu,
                                              bf16* __restrict__ Z){
  __shared__ float rs[256], rq[256];
  int n = blockIdx.x;
  const float* a = attn + (size_t)n*512;
  float v0 = a[threadIdx.x], v1 = a[threadIdx.x + 256];
  rs[threadIdx.x] = v0 + v1; rq[threadIdx.x] = v0*v0 + v1*v1;
  __syncthreads();
  for (int off = 128; off; off >>= 1){
    if (threadIdx.x < off){ rs[threadIdx.x] += rs[threadIdx.x+off]; rq[threadIdx.x] += rq[threadIdx.x+off]; }
    __syncthreads();
  }
  float mu   = rs[0] * (1.f/512.f);
  float var  = rq[0] * (1.f/512.f) - mu*mu;
  float rsig = rsqrtf(var + 1e-5f);
  for (int j = threadIdx.x; j < 512; j += 256){
    int h = j >> 6, d = j & 63;
    float u = (float)qu[(size_t)n*1024 + h*128 + 64 + d];
    Z[(size_t)n*512 + j] = to_bf16((a[j] - mu) * rsig * u);
  }
}

extern "C" void kernel_launch(void* const* d_in, const int* in_sizes, int n_in,
                              void* d_out, int out_size, void* d_ws, size_t ws_size,
                              hipStream_t stream){
  const float* x   = (const float*)d_in[0];
  const float* sub = (const float*)d_in[1];
  const float* Wq  = (const float*)d_in[2];
  const float* Wkv = (const float*)d_in[3];
  const float* Wo  = (const float*)d_in[4];
  const float* bo  = (const float*)d_in[5];
  float* out = (float*)d_out;
  int n = in_sizes[0] / 1024;   // 4096

  char* ws = (char*)d_ws;
  size_t off = 0;
  auto alloc = [&](size_t bytes){ char* p = ws + off; off += (bytes + 255) & ~(size_t)255; return p; };
  bf16*  Wq_b  = (bf16*)alloc((size_t)1024*1024*2);
  bf16*  Wkv_b = (bf16*)alloc((size_t)1024*256*2);
  bf16*  WkT   = (bf16*)alloc((size_t)8*256*64*2);
  bf16*  Wo_b  = (bf16*)alloc((size_t)1024*512*2);
  bf16*  LNx   = (bf16*)alloc((size_t)n*1024*2);
  bf16*  QU    = (bf16*)alloc((size_t)n*1024*2);
  bf16*  G     = (bf16*)alloc((size_t)n*2048*2);
  bf16*  T     = (bf16*)alloc((size_t)n*2048*2);
  float* ATT   = (float*)alloc((size_t)n*512*4);
  bf16*  Z     = (bf16*)alloc((size_t)n*512*2);

  // weight prep (bf16)
  cvt_f32_bf16<<<(1024*1024)/256, 256, 0, stream>>>(Wq,  Wq_b,  1024*1024);
  cvt_f32_bf16<<<(1024*256)/256,  256, 0, stream>>>(Wkv, Wkv_b, 1024*256);
  cvt_f32_bf16<<<(1024*512)/256,  256, 0, stream>>>(Wo,  Wo_b,  1024*512);
  build_wkT<<<(8*256*64)/256, 256, 0, stream>>>(Wkv, WkT);

  // LN(x)
  ln_rows<<<n, 256, 0, stream>>>(x, LNx, 1024);

  // qu = LN(x) @ Wq^T   (M=n, N=1024, K=1024)
  gemm_bf16nt<<<dim3(1024/64, n/64), 256, 0, stream>>>(LNx, 1024, Wq_b, 1024,
                                                       (void*)QU, 1024, 1024, 1, nullptr);
  // g_h = q_h @ Wk_h    (M=n, N=256, K=64), per head
  for (int h = 0; h < 8; ++h)
    gemm_bf16nt<<<dim3(256/64, n/64), 256, 0, stream>>>(QU + h*128, 1024,
                                                        WkT + (size_t)h*256*64, 64,
                                                        (void*)(G + h*256), 2048, 64, 1, nullptr);
  // fused attention core (reads subseq_emb exactly once)
  attn_core<<<n, 256, 0, stream>>>(sub, G, T);

  // attnout_h = t_h @ Wv_h^T  (M=n, N=64, K=256), per head; Wv_h rows are Wkv rows h*128+64..+127
  for (int h = 0; h < 8; ++h)
    gemm_bf16nt<<<dim3(64/64, n/64), 256, 0, stream>>>(T + h*256, 2048,
                                                       Wkv_b + (size_t)(h*128 + 64)*256, 256,
                                                       (void*)(ATT + h*64), 512, 256, 0, nullptr);
  // z = LN(attnout) * u
  make_z<<<n, 256, 0, stream>>>(ATT, QU, Z);

  // out = z @ Wo^T + bo  (M=n, N=1024, K=512), f32 out
  gemm_bf16nt<<<dim3(1024/64, n/64), 256, 0, stream>>>(Z, 512, Wo_b, 512,
                                                       (void*)out, 1024, 512, 0, bo);
}
